// MultiHeadSelfAttention_30502857736619
// MI455X (gfx1250) — matrix operand order repaired and checked by reading, not compiled, not run on hardware
//
#include <hip/hip_runtime.h>
#include <hip/hip_bf16.h>
#include <stdint.h>

// MHSA (heads-crossed variant), B=8 N=4096 C=1024 H=16 D=64, M = B*N = 32768.
// Pipeline: cvt(x->bf16, W->Wt bf16 transposed) -> fused QKV WMMA GEMM (bf16,
// double-buffered async global->LDS) -> per-token WMMA attention -> O GEMM.
// Workspace: assumes ws_size >= ~330 MB.

typedef __bf16 bf16_t;
typedef __attribute__((ext_vector_type(16))) __bf16 v16bf;
typedef __attribute__((ext_vector_type(8)))  __bf16 v8bf;
typedef __attribute__((ext_vector_type(4)))  __bf16 v4bf;
typedef __attribute__((ext_vector_type(8)))  float  v8f;

union BF16x16 { v16bf v; struct { v8bf lo, hi; } p; };

#define M_TOK   32768ull
#define CDIM    1024
#define NHEAD   16
#define DHEAD   64

// ---- workspace offsets (bytes) ----
#define XB_OFF  0ull
#define WT_OFF  (XB_OFF + M_TOK*CDIM*2)
#define Q_OFF   (WT_OFF + 4ull*CDIM*CDIM*2)
#define K_OFF   (Q_OFF + M_TOK*CDIM*2)
#define V_OFF   (K_OFF + M_TOK*CDIM*2)
#define AO_OFF  (V_OFF + M_TOK*CDIM*2)

__device__ __forceinline__ v8f wmma_bf16(v16bf a, v16bf b, v8f c) {
  return __builtin_amdgcn_wmma_f32_16x16x32_bf16(false, a, false, b,
                                                 (short)0, c, false, false);
}

// CDNA5 async global->LDS copy, 16 bytes per lane (tracked by ASYNCcnt).
__device__ __forceinline__ void async_g2l_16B(uint32_t lds_off, const void* g) {
  asm volatile("global_load_async_to_lds_b128 %0, %1, off"
               :: "v"(lds_off), "v"(g) : "memory");
}
__device__ __forceinline__ void async_wait0() {
  asm volatile("s_wait_asynccnt 0" ::: "memory");
}

// ---------------- convert kernels ----------------
__global__ __launch_bounds__(256) void cvt_x(const float* __restrict__ x,
                                             bf16_t* __restrict__ xb) {
  size_t i = ((size_t)blockIdx.x * 256 + threadIdx.x) * 4;
  float4 f = *(const float4*)(x + i);
  v4bf o;
  o[0] = (bf16_t)f.x; o[1] = (bf16_t)f.y; o[2] = (bf16_t)f.z; o[3] = (bf16_t)f.w;
  *(v4bf*)(xb + i) = o;
}

// W[k][n] (f32) -> Wt[n][k] (bf16), 4 matrices via blockIdx.z
__global__ __launch_bounds__(256) void cvt_wt(const float* __restrict__ wq,
                                              const float* __restrict__ wk,
                                              const float* __restrict__ wv,
                                              const float* __restrict__ wo,
                                              bf16_t* __restrict__ out) {
  __shared__ float t[32][33];
  const float* W = (blockIdx.z == 0) ? wq : (blockIdx.z == 1) ? wk
                 : (blockIdx.z == 2) ? wv : wo;
  bf16_t* Wt = out + (size_t)blockIdx.z * CDIM * CDIM;
  int bx = blockIdx.x * 32, by = blockIdx.y * 32;
  int tx = threadIdx.x & 31, ty = threadIdx.x >> 5;
  for (int r = ty; r < 32; r += 8)
    t[r][tx] = W[(size_t)(by + r) * CDIM + bx + tx];
  __syncthreads();
  for (int r = ty; r < 32; r += 8)
    Wt[(size_t)(bx + r) * CDIM + by + tx] = (bf16_t)t[tx][r];
}

// ---------------- bf16 WMMA GEMM: out = A @ Wt^T + bias ----------------
// A: [M,1024] bf16 row-major; Wt: [1024(N),1024(K)] bf16 (transposed weight).
// Block tile 128x256, BK=32, 8 waves (2x4), wave tile 64x64 (4x4 WMMA tiles).
// Double-buffered async-to-LDS pipeline, one barrier per K-step.
template<bool OUT_BF16>
__global__ __launch_bounds__(256) void gemm_bf16_k(
    const bf16_t* __restrict__ A,
    const bf16_t* __restrict__ WtBase,
    const float* __restrict__ b0, const float* __restrict__ b1,
    const float* __restrict__ b2,
    void* __restrict__ outBase) {
  constexpr int LDK = 40;  // padded row stride (halves) to spread LDS banks
  constexpr int A_HALVES = 128 * LDK;   // per buffer
  constexpr int B_HALVES = 256 * LDK;
  __shared__ bf16_t sA[2][A_HALVES];
  __shared__ bf16_t sB[2][B_HALVES];

  const int tid  = threadIdx.x;
  const int lane = tid & 31;
  const int wave = tid >> 5;
  const int l16  = lane & 15;
  const int hq   = lane >> 4;            // half-wave select
  const int wm   = (wave >> 2) * 64;     // wave row offset in block tile
  const int wn   = (wave & 3) * 64;      // wave col offset
  const size_t m0 = (size_t)blockIdx.x * 128;
  const int    n0 = blockIdx.y * 256;
  const int    z  = blockIdx.z;

  const bf16_t* Wt   = WtBase + (size_t)z * (CDIM * CDIM);
  const float*  bias = (z == 0) ? b0 : (z == 1) ? b1 : b2;

  v8f acc[4][4];
  #pragma unroll
  for (int i = 0; i < 4; ++i)
    #pragma unroll
    for (int j = 0; j < 4; ++j)
      #pragma unroll
      for (int q = 0; q < 8; ++q) acc[i][j][q] = 0.0f;

  const uint32_t aBase = (uint32_t)(uintptr_t)&sA[0][0];
  const uint32_t bBase = (uint32_t)(uintptr_t)&sB[0][0];

  // Issue async copies of one K-tile (A:128x32 = 512 x16B, B:256x32 = 1024
  // x16B chunks; 2+4 chunks per thread) into buffer `buf`.
  auto issue_tile = [&](int k0, int buf) {
    const uint32_t aB = aBase + (uint32_t)(buf * A_HALVES * 2);
    const uint32_t bB = bBase + (uint32_t)(buf * B_HALVES * 2);
    #pragma unroll
    for (int p = 0; p < 2; ++p) {
      int chunk = tid + p * 256;          // 0..511
      int row = chunk >> 2, c = chunk & 3;
      async_g2l_16B(aB + (uint32_t)((row * LDK + c * 8) * 2),
                    A + (m0 + row) * CDIM + k0 + c * 8);
    }
    #pragma unroll
    for (int p = 0; p < 4; ++p) {
      int chunk = tid + p * 256;          // 0..1023
      int row = chunk >> 2, c = chunk & 3;
      async_g2l_16B(bB + (uint32_t)((row * LDK + c * 8) * 2),
                    Wt + (size_t)(n0 + row) * CDIM + k0 + c * 8);
    }
  };

  issue_tile(0, 0);

  for (int it = 0; it < CDIM / 32; ++it) {
    async_wait0();        // this wave's copies for tile `it` are in LDS
    __syncthreads();      // all waves' copies visible; prev reads of buf^1 done
    if (it + 1 < CDIM / 32) issue_tile((it + 1) * 32, (it + 1) & 1);

    const bf16_t* sAb = sA[it & 1];
    const bf16_t* sBb = sB[it & 1];

    BF16x16 bfr[4];
    #pragma unroll
    for (int j = 0; j < 4; ++j) {
      const bf16_t* bp = &sBb[(wn + j * 16 + l16) * LDK + hq * 8];
      bfr[j].p.lo = *(const v8bf*)bp;
      bfr[j].p.hi = *(const v8bf*)(bp + 16);
    }
    #pragma unroll
    for (int i = 0; i < 4; ++i) {
      BF16x16 af;
      const bf16_t* ap = &sAb[(wm + i * 16 + l16) * LDK + hq * 8];
      af.p.lo = *(const v8bf*)ap;        // K base..base+7
      af.p.hi = *(const v8bf*)(ap + 16); // K base+16..base+23
      #pragma unroll
      for (int j = 0; j < 4; ++j)
        acc[i][j] = wmma_bf16(af.v, bfr[j].v, acc[i][j]);
    }
  }

  // epilogue: C layout lane l -> (M = q + 8*hq, N = l16) per VGPR q
  #pragma unroll
  for (int j = 0; j < 4; ++j) {
    const int col = n0 + wn + j * 16 + l16;
    const float bval = bias[col];
    #pragma unroll
    for (int i = 0; i < 4; ++i) {
      const size_t row0 = m0 + wm + i * 16 + hq * 8;
      if constexpr (OUT_BF16) {
        bf16_t* out = (bf16_t*)outBase + (size_t)z * (M_TOK * CDIM);
        #pragma unroll
        for (int q = 0; q < 8; ++q)
          out[(row0 + q) * CDIM + col] = (bf16_t)(acc[i][j][q] + bval);
      } else {
        float* out = (float*)outBase;
        #pragma unroll
        for (int q = 0; q < 8; ++q)
          out[(row0 + q) * CDIM + col] = acc[i][j][q] + bval;
      }
    }
  }
}

// ---------------- per-token attention (one wave = one token) ----------------
// energy[16,16] = (q @ k^T) / 8 ; softmax over k-head ; out = attn @ v [16,64]
__global__ __launch_bounds__(256) void attn_kernel(
    const bf16_t* __restrict__ Q, const bf16_t* __restrict__ K,
    const bf16_t* __restrict__ V, bf16_t* __restrict__ AO) {
  __shared__ bf16_t sAttn[8][16][16];
  __shared__ bf16_t sV[8][16][64];

  const int wave = threadIdx.x >> 5;
  const int lane = threadIdx.x & 31;
  const int l16  = lane & 15;
  const int hq   = lane >> 4;
  const size_t t = (size_t)blockIdx.x * 8 + wave;

  const bf16_t* q = Q + t * CDIM;
  const bf16_t* k = K + t * CDIM;
  const bf16_t* v = V + t * CDIM;

  // stage this token's V into LDS (coalesced 16B chunks)
  #pragma unroll
  for (int i = 0; i < 4; ++i) {
    int chunk = lane + 32 * i;  // 128 chunks of 8 halves
    *(v8bf*)&sV[wave][0][chunk * 8] = *(const v8bf*)(v + chunk * 8);
  }

  // energy = q @ k^T over D=64 (two K=32 WMMA steps); rows=q-head, cols=k-head
  v8f e;
  for (int i = 0; i < 8; ++i) e[i] = 0.0f;
  #pragma unroll
  for (int c = 0; c < 2; ++c) {
    BF16x16 af, bfg;
    const bf16_t* qp = q + l16 * DHEAD + c * 32 + hq * 8;
    af.p.lo = *(const v8bf*)qp;
    af.p.hi = *(const v8bf*)(qp + 16);
    const bf16_t* kp = k + l16 * DHEAD + c * 32 + hq * 8;
    bfg.p.lo = *(const v8bf*)kp;
    bfg.p.hi = *(const v8bf*)(kp + 16);
    e = wmma_bf16(af.v, bfg.v, e);
  }

  // softmax across the 16 lanes of each half-wave (N axis), per row q
  #pragma unroll
  for (int j = 0; j < 8; ++j) {
    float xv = e[j] * 0.125f;  // 1/sqrt(64)
    float mx = xv;
    mx = fmaxf(mx, __shfl_xor(mx, 1));
    mx = fmaxf(mx, __shfl_xor(mx, 2));
    mx = fmaxf(mx, __shfl_xor(mx, 4));
    mx = fmaxf(mx, __shfl_xor(mx, 8));
    float ex = __expf(xv - mx);
    float sm = ex;
    sm += __shfl_xor(sm, 1);
    sm += __shfl_xor(sm, 2);
    sm += __shfl_xor(sm, 4);
    sm += __shfl_xor(sm, 8);
    sAttn[wave][j + 8 * hq][l16] = (bf16_t)(ex / sm);
  }
  __syncthreads();

  // out = attn[16x16] @ v[16x64], K padded 16->32 with zeros
  #pragma unroll
  for (int nt = 0; nt < DHEAD; nt += 16) {
    BF16x16 af, bfg;
    af.p.lo = *(const v8bf*)&sAttn[wave][l16][hq * 8];  // K 0..15 real
    #pragma unroll
    for (int i = 0; i < 8; ++i) af.p.hi[i] = (bf16_t)0.0f;  // K 16..31 zero
    for (int j = 0; j < 16; ++j)
      bfg.v[j] = (j >= 8) ? (bf16_t)0.0f : sV[wave][hq * 8 + j][nt + l16];
    v8f o;
    for (int i = 0; i < 8; ++i) o[i] = 0.0f;
    o = wmma_bf16(af.v, bfg.v, o);
    #pragma unroll
    for (int j = 0; j < 8; ++j)
      AO[t * CDIM + (size_t)(j + 8 * hq) * DHEAD + nt + l16] = (bf16_t)o[j];
  }
}

extern "C" void kernel_launch(void* const* d_in, const int* in_sizes, int n_in,
                              void* d_out, int out_size, void* d_ws,
                              size_t ws_size, hipStream_t stream) {
  (void)in_sizes; (void)n_in; (void)out_size; (void)ws_size;
  const float* x  = (const float*)d_in[0];
  const float* Wq = (const float*)d_in[1];
  const float* bq = (const float*)d_in[2];
  const float* Wk = (const float*)d_in[3];
  const float* bk = (const float*)d_in[4];
  const float* Wv = (const float*)d_in[5];
  const float* bv = (const float*)d_in[6];
  const float* Wo = (const float*)d_in[7];
  const float* bo = (const float*)d_in[8];

  char* ws = (char*)d_ws;
  bf16_t* xb  = (bf16_t*)(ws + XB_OFF);
  bf16_t* wt  = (bf16_t*)(ws + WT_OFF);  // [Wq^T, Wk^T, Wv^T, Wo^T]
  bf16_t* Qb  = (bf16_t*)(ws + Q_OFF);
  bf16_t* Kb  = (bf16_t*)(ws + K_OFF);
  bf16_t* Vb  = (bf16_t*)(ws + V_OFF);
  bf16_t* AOb = (bf16_t*)(ws + AO_OFF);

  // 1) precision conversion + weight transpose
  cvt_x<<<32768, 256, 0, stream>>>(x, xb);
  cvt_wt<<<dim3(32, 32, 4), 256, 0, stream>>>(Wq, Wk, Wv, Wo, wt);

  // 2) fused Q/K/V projections: grid.z selects weight/bias/output
  gemm_bf16_k<true><<<dim3(256, 4, 3), 256, 0, stream>>>(
      xb, wt, bq, bk, bv, (void*)Qb);

  // 3) per-token cross-head attention (32768 tokens, 8 waves/block)
  attn_kernel<<<4096, 256, 0, stream>>>(Qb, Kb, Vb, AOb);

  // 4) output projection, f32 result
  gemm_bf16_k<false><<<dim3(256, 4, 1), 256, 0, stream>>>(
      AOb, wt + 3ull * CDIM * CDIM, bo, bo, bo, d_out);
}
